// AllForgetLSTM_5806795784740
// MI455X (gfx1250) — matrix operand order repaired and checked by reading, not compiled, not run on hardware
//
#include <hip/hip_runtime.h>
#include <hip/hip_bf16.h>

// ---------------- problem constants ----------------
#define T_STEPS 512
#define BATCH   64
#define HID     512
#define GATES   2048               // 4*HID
#define KDIM    1024               // HID (input) + HID (recurrent)
#define BH      (BATCH * HID)      // 32768
#define LDA     1032               // padded A row stride (2064B = odd multiple of 16B)
#define GSTRIDE 20                 // padded gate-tile row stride (floats)

// d_out layout: out1 [T,B,H] | hs [2,B,H] | cs [2,B,H]
#define HS_OFF  ((size_t)T_STEPS * BH)      // 16777216
#define CS_OFF  (HS_OFF + 2 * (size_t)BH)

// ---------------- workspace layout (bytes) ----------------
#define OFF_BAR    ((size_t)0)
#define OFF_WCAT0  ((size_t)4096)
#define OFF_WCAT1  (OFF_WCAT0 + (size_t)GATES * KDIM * 2)    // +4 MB
#define OFF_BIAS0  (OFF_WCAT1 + (size_t)GATES * KDIM * 2)    // +4 MB
#define OFF_BIAS1  (OFF_BIAS0 + (size_t)GATES * 4)
#define OFF_XBF    (OFF_BIAS1 + (size_t)GATES * 4)           // +32 MB  (x in bf16)
#define OFF_OUT0   (OFF_XBF + (size_t)T_STEPS * BH * 2)      // +32 MB  (layer0 h seq, bf16)
#define OFF_HZERO  (OFF_OUT0 + (size_t)T_STEPS * BH * 2)
#define OFF_HBUF1  (OFF_HZERO + (size_t)BH * 2)
#define OFF_C0     (OFF_HBUF1 + (size_t)2 * BH * 2)
#define OFF_C1     (OFF_C0 + (size_t)BH * 4)
#define WS_TOTAL   (OFF_C1 + (size_t)BH * 4)

typedef __attribute__((ext_vector_type(16))) __bf16 v16bf;
typedef __attribute__((ext_vector_type(8)))  __bf16 v8bf;
typedef __attribute__((ext_vector_type(8)))  float  v8f;
typedef int v4i_ __attribute__((vector_size(16)));   // type the async builtin wants

__device__ __forceinline__ unsigned short f32_to_bf16(float f) {
  unsigned u = __float_as_uint(f);
  unsigned r = u + 0x7FFFu + ((u >> 16) & 1u);   // round-to-nearest-even
  return (unsigned short)(r >> 16);
}

__device__ __forceinline__ float sigmoidf_(float x) {
  return 1.0f / (1.0f + __expf(-x));
}

// ---- CDNA5 async global->LDS copy (ASYNCcnt path), with sync fallback ----
#if defined(__gfx1250__) && __has_builtin(__builtin_amdgcn_global_load_async_to_lds_b128)
#define HAVE_ASYNC_LDS 1
#endif

__device__ __forceinline__ void async_copy_b128(const unsigned short* gsrc,
                                                unsigned short* ldst) {
#ifdef HAVE_ASYNC_LDS
  typedef __attribute__((address_space(3))) v4i_* las_t;
  v4i_* g = (v4i_*)gsrc;                                   // generic ptr to 16B
  las_t  l = (las_t)(unsigned)(size_t)ldst;                // generic->LDS (low 32 bits)
  __builtin_amdgcn_global_load_async_to_lds_b128(g, l, 0, 0);
#else
  const uint4 v = *(const uint4*)gsrc;
  *(uint4*)ldst = v;
#endif
}

__device__ __forceinline__ void wait_async_copies() {
#ifdef HAVE_ASYNC_LDS
#if __has_builtin(__builtin_amdgcn_s_wait_asynccnt)
  __builtin_amdgcn_s_wait_asynccnt(0);
#else
  asm volatile("s_wait_asynccnt 0" ::: "memory");
#endif
#endif
}

// sense-reversing device-wide barrier (32 co-resident workgroups)
__device__ __forceinline__ void grid_sync(unsigned* cnt, unsigned* gen, unsigned nblk) {
  __syncthreads();
  if (threadIdx.x == 0) {
    unsigned g = __hip_atomic_load(gen, __ATOMIC_RELAXED, __HIP_MEMORY_SCOPE_AGENT);
    unsigned a = __hip_atomic_fetch_add(cnt, 1u, __ATOMIC_ACQ_REL, __HIP_MEMORY_SCOPE_AGENT);
    if (a == nblk - 1u) {
      __hip_atomic_store(cnt, 0u, __ATOMIC_RELAXED, __HIP_MEMORY_SCOPE_AGENT);
      __hip_atomic_fetch_add(gen, 1u, __ATOMIC_ACQ_REL, __HIP_MEMORY_SCOPE_AGENT);
    } else {
      while (__hip_atomic_load(gen, __ATOMIC_ACQUIRE, __HIP_MEMORY_SCOPE_AGENT) == g) {
        __builtin_amdgcn_s_sleep(1);
      }
    }
  }
  __syncthreads();
}

// ---------------- setup kernels ----------------
__global__ void zero_kernel(unsigned* p, int n) {
  int i = blockIdx.x * blockDim.x + threadIdx.x;
  if (i < n) p[i] = 0u;
}

__global__ void build_xbf_kernel(const float* __restrict__ src,
                                 unsigned short* __restrict__ dst, int n) {
  int i = blockIdx.x * blockDim.x + threadIdx.x;
  if (i < n) dst[i] = f32_to_bf16(src[i]);
}

// Wcat[g][k] = k<HID ? W_ih[g][k] : W_hh[g][k-HID], converted to bf16
__global__ void build_wcat_kernel(const float* __restrict__ wih,
                                  const float* __restrict__ whh,
                                  unsigned short* __restrict__ dst) {
  int i = blockIdx.x * blockDim.x + threadIdx.x;
  if (i < GATES * KDIM) {
    int g = i >> 10;
    int k = i & (KDIM - 1);
    float v = (k < HID) ? wih[(size_t)g * HID + k] : whh[(size_t)g * HID + (k - HID)];
    dst[i] = f32_to_bf16(v);
  }
}

__global__ void build_bias_kernel(const float* __restrict__ bi,
                                  const float* __restrict__ bh,
                                  float* __restrict__ dst) {
  int i = blockIdx.x * blockDim.x + threadIdx.x;
  if (i < GATES) dst[i] = bi[i] + bh[i];
}

// ---------------- persistent LSTM layer kernel ----------------
// 32 workgroups x 256 threads. WG owns H-columns [16*blockIdx.x, +16).
// Per step: stage the FULL A operand [x_t | h_{t-1}] (64 x 1024 bf16, 129KB)
// into LDS via async global->LDS copies, then each wave runs 64 back-to-back
// v_wmma_f32_16x16x32_bf16 (wave w -> gate w>>1, m-tiles {2*(w&1),2*(w&1)+1}),
// then the i/f/g/o elementwise is fused locally through LDS.
template <int LAYER>
__global__ void lstm_seq_kernel(const unsigned short* __restrict__ xin_bf,  // [T][64][512] bf16
                                const unsigned short* __restrict__ Wcat,    // [2048][1024] bf16
                                const float* __restrict__ bias,             // [2048]
                                unsigned short* __restrict__ out0bf,        // LAYER==0 seq out (bf16)
                                unsigned short* __restrict__ hbuf,          // LAYER==1 h double buffer
                                const unsigned short* __restrict__ hzero,   // zeros [64][512]
                                float* __restrict__ cbuf,                   // [64][512] f32
                                float* __restrict__ out_f32,                // d_out
                                unsigned* __restrict__ bar_cnt,
                                unsigned* __restrict__ bar_gen) {
  extern __shared__ __align__(16) char smem[];
  unsigned short* ldsA = (unsigned short*)smem;               // [64][LDA]
  float* ldsG = (float*)(smem + (size_t)64 * LDA * 2);        // [4][64][GSTRIDE]

  const int tid    = threadIdx.x;
  const int wave   = tid >> 5;
  const int lane   = tid & 31;
  const int gate   = wave >> 1;             // 0..3
  const int mt0    = (wave & 1) * 2;        // m-tile pair base
  const int laneM  = lane & 15;
  const int hiHalf = lane >> 4;             // 0 or 1
  const int ncol0  = blockIdx.x * 16;       // H-column slice base

  // B fragment source: row (gate*512 + ncol0 + laneM) of Wcat, contiguous in K
  const unsigned short* Brow =
      Wcat + ((size_t)(gate * HID + ncol0 + laneM)) * KDIM;

  // staging: 256 threads -> 64 rows x (4 threads/row), 16B chunks
  const int srow = tid >> 2;
  const int sq   = tid & 3;
  unsigned short* rowL = ldsA + (size_t)srow * LDA;

  const unsigned short* rA0 = ldsA + (size_t)(mt0 * 16 + laneM) * LDA;
  const unsigned short* rA1 = rA0 + (size_t)16 * LDA;

  #pragma unroll 1
  for (int t = 0; t < T_STEPS; ++t) {
    const unsigned short* hprev;
    unsigned short* hnext;
    if (LAYER == 0) {
      hprev = (t == 0) ? hzero : (out0bf + (size_t)(t - 1) * BH);
      hnext = out0bf + (size_t)t * BH;
    } else {
      hprev = hbuf + (size_t)(t & 1) * BH;
      hnext = hbuf + (size_t)((t + 1) & 1) * BH;
    }

    // ---- stage full A = [x_t | h_{t-1}] into LDS (async global->LDS) ----
    {
      const unsigned short* rowX = xin_bf + ((size_t)t * BATCH + srow) * HID;
      const unsigned short* rowH = hprev + (size_t)srow * HID;
      #pragma unroll
      for (int ch = 0; ch < 16; ++ch) {
        const int cidx = sq + ch * 4;                       // 0..63 -> 8-elem chunks
        async_copy_b128(rowX + cidx * 8, rowL + cidx * 8);
      }
      #pragma unroll
      for (int ch = 0; ch < 16; ++ch) {
        const int cidx = sq + ch * 4;
        async_copy_b128(rowH + cidx * 8, rowL + HID + cidx * 8);
      }
      wait_async_copies();
      __syncthreads();   // A chunk visible to all waves
    }

    // ---- 32 k-steps of K=32 WMMA, uninterrupted ----
    v8f acc0 = {};
    v8f acc1 = {};
    #pragma unroll 4
    for (int ks = 0; ks < 32; ++ks) {
      const int lo = ks * 32 + hiHalf * 8;     // 16-bit A 16x32 lane layout
      union { v16bf v; v8bf h[2]; } a0, a1;
      a0.h[0] = *(const v8bf*)(rA0 + lo);
      a0.h[1] = *(const v8bf*)(rA0 + lo + 16);
      a1.h[0] = *(const v8bf*)(rA1 + lo);
      a1.h[1] = *(const v8bf*)(rA1 + lo + 16);

      union { v16bf v; v8bf h[2]; } bu;
      bu.h[0] = *(const v8bf*)(Brow + lo);
      bu.h[1] = *(const v8bf*)(Brow + lo + 16);
      const v16bf bf = bu.v;  // B 32x16

      acc0 = __builtin_amdgcn_wmma_f32_16x16x32_bf16(
          false, a0.v, false, bf, (short)0, acc0, false, false);
      acc1 = __builtin_amdgcn_wmma_f32_16x16x32_bf16(
          false, a1.v, false, bf, (short)0, acc1, false, false);
    }

    // ---- accumulators -> gate LDS (C/D layout: M = r + 8*hiHalf, N = laneM) ----
    {
      const int rb = hiHalf * 8;
      #pragma unroll
      for (int r = 0; r < 8; ++r) {
        ldsG[((gate * 64) + mt0 * 16 + rb + r) * GSTRIDE + laneM]      = acc0[r];
        ldsG[((gate * 64) + mt0 * 16 + 16 + rb + r) * GSTRIDE + laneM] = acc1[r];
      }
    }
    __syncthreads();

    // ---- fused elementwise: 64 x 16 cells for this WG's slice ----
    #pragma unroll
    for (int it = 0; it < 4; ++it) {
      const int e = tid + it * 256;
      const int b = e >> 4;
      const int n = e & 15;
      const int col = ncol0 + n;
      float iv = sigmoidf_(ldsG[(0 * 64 + b) * GSTRIDE + n] + bias[col]);
      float fv = sigmoidf_(ldsG[(1 * 64 + b) * GSTRIDE + n] + bias[HID + col]);
      float gv = tanhf(ldsG[(2 * 64 + b) * GSTRIDE + n] + bias[2 * HID + col]);
      float ov = sigmoidf_(ldsG[(3 * 64 + b) * GSTRIDE + n] + bias[3 * HID + col]);
      const size_t ci = (size_t)b * HID + col;
      float cn = fv * cbuf[ci] + iv * gv;
      cbuf[ci] = cn;
      float h = ov * tanhf(cn);
      hnext[ci] = f32_to_bf16(h);
      if (LAYER == 1) out_f32[((size_t)t * BATCH + b) * HID + col] = h;
      if (t == T_STEPS - 1) {
        out_f32[HS_OFF + (size_t)LAYER * BH + ci] = h;
        out_f32[CS_OFF + (size_t)LAYER * BH + ci] = cn;
      }
    }

    __threadfence();                       // publish h/c before step barrier
    grid_sync(bar_cnt, bar_gen, gridDim.x);
  }
}

// ---------------- host launch ----------------
extern "C" void kernel_launch(void* const* d_in, const int* in_sizes, int n_in,
                              void* d_out, int out_size, void* d_ws, size_t ws_size,
                              hipStream_t stream) {
  const float* x    = (const float*)d_in[0];
  const float* Wih0 = (const float*)d_in[1];
  const float* Whh0 = (const float*)d_in[2];
  const float* bih0 = (const float*)d_in[3];
  const float* bhh0 = (const float*)d_in[4];
  const float* Wih1 = (const float*)d_in[5];
  const float* Whh1 = (const float*)d_in[6];
  const float* bih1 = (const float*)d_in[7];
  const float* bhh1 = (const float*)d_in[8];

  char* ws = (char*)d_ws;
  unsigned*       bar    = (unsigned*)(ws + OFF_BAR);
  unsigned short* wcat0  = (unsigned short*)(ws + OFF_WCAT0);
  unsigned short* wcat1  = (unsigned short*)(ws + OFF_WCAT1);
  float*          bias0  = (float*)(ws + OFF_BIAS0);
  float*          bias1  = (float*)(ws + OFF_BIAS1);
  unsigned short* xbf    = (unsigned short*)(ws + OFF_XBF);
  unsigned short* out0bf = (unsigned short*)(ws + OFF_OUT0);
  unsigned short* hzero  = (unsigned short*)(ws + OFF_HZERO);
  unsigned short* hbuf1  = (unsigned short*)(ws + OFF_HBUF1);
  float*          c0     = (float*)(ws + OFF_C0);
  float*          c1     = (float*)(ws + OFF_C1);
  float*          out    = (float*)d_out;

  // zero barrier + hzero + hbuf1 + c0 + c1 (contiguous from OFF_HZERO)
  zero_kernel<<<1, 64, 0, stream>>>(bar, 64);
  {
    int nwords = (int)((WS_TOTAL - OFF_HZERO) / 4);
    zero_kernel<<<(nwords + 255) / 256, 256, 0, stream>>>(
        (unsigned*)(ws + OFF_HZERO), nwords);
  }

  // one-shot conversions: x -> bf16, concatenated bf16 weights, fused biases
  build_xbf_kernel<<<(T_STEPS * BH + 255) / 256, 256, 0, stream>>>(
      x, xbf, T_STEPS * BH);
  build_wcat_kernel<<<(GATES * KDIM + 255) / 256, 256, 0, stream>>>(Wih0, Whh0, wcat0);
  build_wcat_kernel<<<(GATES * KDIM + 255) / 256, 256, 0, stream>>>(Wih1, Whh1, wcat1);
  build_bias_kernel<<<(GATES + 255) / 256, 256, 0, stream>>>(bih0, bhh0, bias0);
  build_bias_kernel<<<(GATES + 255) / 256, 256, 0, stream>>>(bih1, bhh1, bias1);

  const size_t smem_bytes = (size_t)64 * LDA * 2 + (size_t)4 * 64 * GSTRIDE * 4;  // 152576

  // layer 0: bf16 x -> bf16 sequence out0bf (+ final states)
  lstm_seq_kernel<0><<<dim3(32), dim3(256), smem_bytes, stream>>>(
      xbf, wcat0, bias0, out0bf, nullptr, hzero, c0, out, bar, bar + 1);

  // layer 1: bf16 out0bf -> f32 out1 in d_out (+ final states)
  lstm_seq_kernel<1><<<dim3(32), dim3(256), smem_bytes, stream>>>(
      out0bf, wcat1, bias1, nullptr, hbuf1, hzero, c1, out, bar, bar + 1);
}
